// HSTUModel_73229192397476
// MI455X (gfx1250) — compile-verified, matrix operand order checked by reading, not run on hardware
//
#include <hip/hip_runtime.h>

#define B_   8
#define N_   1025
#define D_   1024
#define H_   8
#define HD_  128
#define L_   2
#define M_   (B_ * N_)      // 8200
#define FOURD (4 * D_)      // 4096
#define TWOD  (2 * D_)      // 2048
#define NBUCK 64

typedef __attribute__((ext_vector_type(16))) _Float16 v16h;
typedef __attribute__((ext_vector_type(8)))  float    v8f;

struct __align__(16) U4 { unsigned int x, y, z, w; };

__device__ __forceinline__ float silu_f(float x) { return x / (1.f + expf(-x)); }

__device__ __forceinline__ v8f zero8() {
    v8f c = {0.f, 0.f, 0.f, 0.f, 0.f, 0.f, 0.f, 0.f};
    return c;
}

__device__ __forceinline__ v16h frag_zero16() {
    v16h a;
    U4 z{0u, 0u, 0u, 0u};
    ((U4*)&a)[0] = z; ((U4*)&a)[1] = z;
    return a;
}

// Load a 16x32 f16 fragment (A or Bt layout) from a row-major tile.
// Lane L<16: row L, halves [ko..ko+7] and [ko+16..ko+23] with ko=0;
// Lane L>=16: same row, ko=8.  Matches CDNA5 16-bit A/B VGPR striping.
__device__ __forceinline__ v16h load_frag(const _Float16* base, int ld, int lane) {
    int m  = lane & 15;
    int ko = (lane >> 4) << 3;
    const _Float16* p = base + m * ld + ko;
    v16h a;
    ((U4*)&a)[0] = *(const U4*)(p);
    ((U4*)&a)[1] = *(const U4*)(p + 16);
    return a;
}

__device__ __forceinline__ v16h load_frag_g_guard(const _Float16* base, long ld, int lane, int rows_valid) {
    int m  = lane & 15;
    int ko = (lane >> 4) << 3;
    if (m < rows_valid) {
        const _Float16* p = base + (long)m * ld + ko;
        v16h a;
        ((U4*)&a)[0] = *(const U4*)(p);
        ((U4*)&a)[1] = *(const U4*)(p + 16);
        return a;
    }
    return frag_zero16();
}

__device__ __forceinline__ v8f wmma_f16(v16h a, v16h b, v8f c) {
    return __builtin_amdgcn_wmma_f32_16x16x32_f16(false, a, false, b, (short)0, c, false, false);
}

// ---------------------------------------------------------------------------
// Generic f16 GEMM: C[M,Nc] = A[M,K] * B[K,Nc], fused epilogues.
// Ping-pong double-buffered LDS: global->reg prefetch of tile t+1 is issued
// before the 8 WMMAs of tile t so HBM/L2 latency overlaps matrix math.
// mode 0: out16 = silu(acc)
// mode 1: out16 = acc + bias[col]
// mode 2: outf  = acc + bias[col] + res[row,col]
// ---------------------------------------------------------------------------
#define GT_M 128
#define GT_N 128
#define GT_K 32
#define GLD  40   // padded LDS row stride (halves); 80B rows keep 16B alignment

__global__ __launch_bounds__(256)
void gemm_f16_kernel(const _Float16* __restrict__ A, const _Float16* __restrict__ Bw,
                     _Float16* __restrict__ out16, float* __restrict__ outf,
                     const float* __restrict__ bias, const float* __restrict__ res,
                     int M, int K, int Nc, int mode) {
    __shared__ _Float16 As[2][GT_M * GLD];
    __shared__ _Float16 Bs[2][GT_N * GLD];   // stored transposed: Bs[buf][n][k]

    const int tid  = threadIdx.x;
    const int lane = tid & 31;
    const int wid  = tid >> 5;
    const int bm   = blockIdx.y * GT_M;
    const int bn   = blockIdx.x * GT_N;
    const int wm   = (wid & 3) * 32;
    const int wn   = (wid >> 2) * 64;

    v8f acc[2][4];
#pragma unroll
    for (int mf = 0; mf < 2; ++mf)
#pragma unroll
        for (int nf = 0; nf < 4; ++nf) acc[mf][nf] = zero8();

    const int arow = tid >> 1;
    const int ac0  = (tid & 1) * 16;
    const int bkk  = tid >> 3;
    const int bn0  = (tid & 7) * 16;
    const int gm   = bm + arow;
    const bool a_ok = (gm < M);
    const _Float16* aptr = A + (long)gm * K + ac0;               // + kt
    const _Float16* bptr = Bw + (long)bkk * Nc + bn + bn0;       // + kt*Nc

    U4 ra0{0u,0u,0u,0u}, ra1{0u,0u,0u,0u}, rb0, rb1;

    // ---- stage-in tile 0 ----
    {
        if (a_ok) { ra0 = *(const U4*)(aptr); ra1 = *(const U4*)(aptr + 8); }
        rb0 = *(const U4*)(bptr);
        rb1 = *(const U4*)(bptr + 8);
        *(U4*)&As[0][arow * GLD + ac0]     = ra0;
        *(U4*)&As[0][arow * GLD + ac0 + 8] = ra1;
        const _Float16* e0 = (const _Float16*)&rb0;
        const _Float16* e1 = (const _Float16*)&rb1;
#pragma unroll
        for (int i = 0; i < 8; ++i) {
            Bs[0][(bn0 + i)     * GLD + bkk] = e0[i];
            Bs[0][(bn0 + 8 + i) * GLD + bkk] = e1[i];
        }
    }
    __syncthreads();

    const int nk = K / GT_K;
    for (int t = 0; t < nk; ++t) {
        const int cur = t & 1;
        const bool more = (t + 1 < nk);

        // ---- issue global loads for tile t+1 (overlap with WMMA below) ----
        if (more) {
            const long kt1 = (long)(t + 1) * GT_K;
            ra0 = U4{0u,0u,0u,0u}; ra1 = U4{0u,0u,0u,0u};
            if (a_ok) { ra0 = *(const U4*)(aptr + kt1); ra1 = *(const U4*)(aptr + kt1 + 8); }
            rb0 = *(const U4*)(bptr + kt1 * Nc);
            rb1 = *(const U4*)(bptr + kt1 * Nc + 8);
            if (t + 2 < nk) {   // gfx1250 global_prefetch_b8 for tile t+2
                const long kt2 = (long)(t + 2) * GT_K;
                if (a_ok) __builtin_prefetch(aptr + kt2, 0, 1);
                __builtin_prefetch(bptr + kt2 * Nc, 0, 1);
            }
        }

        // ---- compute tile t from LDS ----
        v16h af[2], bf[4];
#pragma unroll
        for (int mf = 0; mf < 2; ++mf) af[mf] = load_frag(&As[cur][(wm + mf * 16) * GLD], GLD, lane);
#pragma unroll
        for (int nf = 0; nf < 4; ++nf) bf[nf] = load_frag(&Bs[cur][(wn + nf * 16) * GLD], GLD, lane);
#pragma unroll
        for (int mf = 0; mf < 2; ++mf)
#pragma unroll
            for (int nf = 0; nf < 4; ++nf)
                acc[mf][nf] = wmma_f16(af[mf], bf[nf], acc[mf][nf]);

        // ---- store tile t+1 into the other buffer ----
        if (more) {
            const int nxt = cur ^ 1;
            *(U4*)&As[nxt][arow * GLD + ac0]     = ra0;
            *(U4*)&As[nxt][arow * GLD + ac0 + 8] = ra1;
            const _Float16* e0 = (const _Float16*)&rb0;
            const _Float16* e1 = (const _Float16*)&rb1;
#pragma unroll
            for (int i = 0; i < 8; ++i) {
                Bs[nxt][(bn0 + i)     * GLD + bkk] = e0[i];
                Bs[nxt][(bn0 + 8 + i) * GLD + bkk] = e1[i];
            }
        }
        __syncthreads();
    }

    // ---- epilogue: C layout -> lane col = lane&15, rows = (lane>=16?8:0)+r ----
    const int colj  = lane & 15;
    const int rbase = (lane >> 4) << 3;
#pragma unroll
    for (int mf = 0; mf < 2; ++mf) {
#pragma unroll
        for (int nf = 0; nf < 4; ++nf) {
            int col = bn + wn + nf * 16 + colj;
#pragma unroll
            for (int r = 0; r < 8; ++r) {
                int gmr = bm + wm + mf * 16 + rbase + r;
                if (gmr < M) {
                    float v = acc[mf][nf][r];
                    long oidx = (long)gmr * Nc + col;
                    if (mode == 0) {
                        out16[oidx] = (_Float16)silu_f(v);
                    } else if (mode == 1) {
                        out16[oidx] = (_Float16)(v + bias[col]);
                    } else {
                        outf[oidx] = v + bias[col] + res[oidx];
                    }
                }
            }
        }
    }
}

// ---------------------------------------------------------------------------
// Fused attention: one wave per (b, h, 16-row i tile).
// scores = q@k^T + bias ; attn = silu(scores)/N * causal ; av += attn@v
// ---------------------------------------------------------------------------
__global__ __launch_bounds__(32)
void attention_kernel(const _Float16* __restrict__ qkvu16, const float* __restrict__ biasB,
                      float* __restrict__ av) {
    __shared__ _Float16 attnT[16 * 32];     // attn tile, row-major [i_local][j_local]
    __shared__ _Float16 vT[128 * 32];       // vT[d][j_local] = v[j0+j_local][d]

    const int lane = threadIdx.x;
    const int i0 = blockIdx.x * 16;
    const int h  = blockIdx.y;
    const int b  = blockIdx.z;
    const long ld = FOURD;

    const _Float16* qbase = qkvu16 + ((long)(b * N_ + i0)) * ld + h * HD_;
    const _Float16* kbase = qkvu16 + ((long)b * N_) * ld + D_     + h * HD_;
    const _Float16* vbase = qkvu16 + ((long)b * N_) * ld + 2 * D_ + h * HD_;

    const int qrows = (N_ - i0 < 16) ? (N_ - i0) : 16;
    v16h qf[4];
#pragma unroll
    for (int kk = 0; kk < 4; ++kk)
        qf[kk] = load_frag_g_guard(qbase + kk * 32, ld, lane, qrows);

    v8f avf[8];
#pragma unroll
    for (int nf = 0; nf < 8; ++nf) avf[nf] = zero8();

    const int colj  = lane & 15;
    const int rbase = (lane >> 4) << 3;
    const int iend  = (i0 + 15 < N_ - 1) ? (i0 + 15) : (N_ - 1);

    for (int j0 = 0; j0 <= iend; j0 += 32) {
        // prefetch next j-tile's k and v rows (one row per lane)
        {
            int jn = j0 + 32 + lane;
            if (jn <= iend) {
                __builtin_prefetch(kbase + (long)jn * ld, 0, 1);
                __builtin_prefetch(vbase + (long)jn * ld, 0, 1);
            }
        }

        // ---- scores for j0..j0+31 (two 16-col C frags, K=128 -> 4 WMMA each) ----
        v8f sc[2];
#pragma unroll
        for (int jc = 0; jc < 2; ++jc) {
            int jb = j0 + jc * 16;
            int kr = N_ - jb; kr = kr < 0 ? 0 : (kr > 16 ? 16 : kr);
            v8f s = zero8();
#pragma unroll
            for (int kk = 0; kk < 4; ++kk) {
                // B-fragment of k^T chunk == row-major read of k rows jb..jb+15
                v16h kf = load_frag_g_guard(kbase + (long)jb * ld + kk * 32, ld, lane, kr);
                s = wmma_f16(qf[kk], kf, s);
            }
            sc[jc] = s;
        }

        // ---- stage v transposed: vT[d][jj] = v[j0+jj][d] ----
        {
            int j = j0 + lane;
            const _Float16* vp = vbase + (long)j * ld;
            bool ok = (j < N_);
#pragma unroll
            for (int d0 = 0; d0 < HD_; d0 += 8) {
                U4 u{0u,0u,0u,0u};
                if (ok) u = *(const U4*)(vp + d0);
                const _Float16* e = (const _Float16*)&u;
#pragma unroll
                for (int t = 0; t < 8; ++t)
                    vT[(d0 + t) * 32 + lane] = e[t];
            }
        }

        // ---- score epilogue: + bias, silu, /N, causal mask -> attnT (f16) ----
#pragma unroll
        for (int jc = 0; jc < 2; ++jc) {
            int j = j0 + jc * 16 + colj;
#pragma unroll
            for (int r = 0; r < 8; ++r) {
                int i = i0 + rbase + r;
                float aout = 0.f;
                if (i < N_ && j < N_ && j <= i) {
                    float s = sc[jc][r] + biasB[((long)b * N_ + i) * N_ + j];
                    aout = silu_f(s) * (1.f / (float)N_);
                }
                attnT[(rbase + r) * 32 + jc * 16 + colj] = (_Float16)aout;
            }
        }

        // LDS stores (cross-lane) must complete before fragment re-reads
        asm volatile("s_wait_dscnt 0" ::: "memory");

        // ---- av += attn @ v ----
        v16h af = load_frag(attnT, 32, lane);
#pragma unroll
        for (int nf = 0; nf < 8; ++nf) {
            v16h vf = load_frag(&vT[nf * 16 * 32], 32, lane);
            avf[nf] = wmma_f16(af, vf, avf[nf]);
        }
        asm volatile("s_wait_dscnt 0" ::: "memory");
    }

    // ---- store av [16 x 128] ----
#pragma unroll
    for (int nf = 0; nf < 8; ++nf) {
        int d = h * HD_ + nf * 16 + colj;
#pragma unroll
        for (int r = 0; r < 8; ++r) {
            int i = i0 + rbase + r;
            if (i < N_)
                av[((long)(b * N_ + i)) * D_ + d] = avf[nf][r];
        }
    }
}

// ---------------------------------------------------------------------------
// LayerNorm / RMSNorm over D=1024, output f16. One block (256 thr) per row.
// ---------------------------------------------------------------------------
__global__ __launch_bounds__(256)
void norm_f16_kernel(const float* __restrict__ x, _Float16* __restrict__ y, int rms) {
    __shared__ float s1[256];
    __shared__ float s2[256];
    const int row = blockIdx.x;
    const int tid = threadIdx.x;
    const float* xr = x + (long)row * D_;
    float a = 0.f, b = 0.f;
    for (int c = tid; c < D_; c += 256) { float v = xr[c]; a += v; b += v * v; }
    s1[tid] = a; s2[tid] = b;
    __syncthreads();
    for (int off = 128; off > 0; off >>= 1) {
        if (tid < off) { s1[tid] += s1[tid + off]; s2[tid] += s2[tid + off]; }
        __syncthreads();
    }
    float mean = s1[0] * (1.f / D_);
    float ms   = s2[0] * (1.f / D_);
    float mu, sc;
    if (rms) { mu = 0.f; sc = rsqrtf(ms + 1e-8f); }
    else     { mu = mean; sc = rsqrtf(ms - mean * mean + 1e-8f); }
    for (int c = tid; c < D_; c += 256)
        y[(long)row * D_ + c] = (_Float16)((xr[c] - mu) * sc);
}

// y16 = u * layernorm(av)   (u lives in qkvu activation cols [3D, 4D))
__global__ __launch_bounds__(256)
void uln_kernel(const float* __restrict__ av, const _Float16* __restrict__ act16,
                _Float16* __restrict__ y) {
    __shared__ float s1[256];
    __shared__ float s2[256];
    const int row = blockIdx.x;
    const int tid = threadIdx.x;
    const float* xr = av + (long)row * D_;
    float a = 0.f, b = 0.f;
    for (int c = tid; c < D_; c += 256) { float v = xr[c]; a += v; b += v * v; }
    s1[tid] = a; s2[tid] = b;
    __syncthreads();
    for (int off = 128; off > 0; off >>= 1) {
        if (tid < off) { s1[tid] += s1[tid + off]; s2[tid] += s2[tid + off]; }
        __syncthreads();
    }
    float mean = s1[0] * (1.f / D_);
    float var  = s2[0] * (1.f / D_) - mean * mean;
    float sc = rsqrtf(var + 1e-8f);
    const _Float16* ur = act16 + (long)row * FOURD + 3 * D_;
    for (int c = tid; c < D_; c += 256)
        y[(long)row * D_ + c] = (_Float16)((float)ur[c] * (xr[c] - mean) * sc);
}

// bias[b,i,j] = pos_w[j-i+N-1] + ts_w[bucket(|ts[b,min(i+1,N-1)] - ts[b,j]|)]
__global__ __launch_bounds__(256)
void bias_kernel(const long long* __restrict__ ts, const float* __restrict__ tsw,
                 const float* __restrict__ posw, float* __restrict__ biasB) {
    long idx = (long)blockIdx.x * 256 + threadIdx.x;
    const long tot = (long)B_ * N_ * N_;
    if (idx >= tot) return;
    int j = (int)(idx % N_);
    long t = idx / N_;
    int i = (int)(t % N_);
    int b = (int)(t / N_);
    float pos = posw[j - i + N_ - 1];
    int i1 = (i + 1 < N_) ? (i + 1) : (N_ - 1);
    long long d = ts[(long)b * N_ + i1] - ts[(long)b * N_ + j];
    if (d < 0) d = -d;
    if (d < 1) d = 1;
    int bucket = (int)(logf((float)d) / 0.301f);
    bucket = bucket < 0 ? 0 : (bucket > NBUCK ? NBUCK : bucket);
    biasB[idx] = pos + tsw[bucket];
}

// p = silu(g) * vv  where vv = vg[:, :2D], g = vg[:, 2D:]
__global__ __launch_bounds__(256)
void swiglu_kernel(const _Float16* __restrict__ vg, _Float16* __restrict__ p) {
    long i = (long)blockIdx.x * 256 + threadIdx.x;
    const long tot = (long)M_ * TWOD;
    if (i >= tot) return;
    long m = i / TWOD;
    int  c = (int)(i % TWOD);
    float vv = (float)vg[m * FOURD + c];
    float g  = (float)vg[m * FOURD + TWOD + c];
    p[i] = (_Float16)(silu_f(g) * vv);
}

__global__ __launch_bounds__(256)
void convert_f16_kernel(const float* __restrict__ src, _Float16* __restrict__ dst, long n) {
    long i = (long)blockIdx.x * 256 + threadIdx.x;
    if (i < n) dst[i] = (_Float16)src[i];
}

// ---------------------------------------------------------------------------
extern "C" void kernel_launch(void* const* d_in, const int* in_sizes, int n_in,
                              void* d_out, int out_size, void* d_ws, size_t ws_size,
                              hipStream_t stream) {
    (void)in_sizes; (void)n_in; (void)out_size;

    const float*     seqs  = (const float*)d_in[0];
    const long long* ats   = (const long long*)d_in[2];
    const float*     qkvuW = (const float*)d_in[3];
    const float*     outW  = (const float*)d_in[4];
    const float*     outBv = (const float*)d_in[5];
    const float*     tsW   = (const float*)d_in[6];
    const float*     posW  = (const float*)d_in[7];
    const float*     fc1W  = (const float*)d_in[8];
    const float*     fc1Bv = (const float*)d_in[9];
    const float*     fc2W  = (const float*)d_in[10];
    const float*     fc2Bv = (const float*)d_in[11];
    float* out = (float*)d_out;
    char*  ws  = (char*)d_ws;

    size_t o = 0;
    auto take = [&](size_t bytes) { size_t r = o; o += (bytes + 255) & ~(size_t)255; return r; };
    _Float16* qkvu16 = (_Float16*)(ws + take((size_t)L_ * D_ * FOURD * 2));
    _Float16* out16w = (_Float16*)(ws + take((size_t)L_ * D_ * D_ * 2));
    _Float16* fc1_16 = (_Float16*)(ws + take((size_t)L_ * D_ * FOURD * 2));
    _Float16* fc2_16 = (_Float16*)(ws + take((size_t)L_ * TWOD * D_ * 2));
    _Float16* Xn16   = (_Float16*)(ws + take((size_t)M_ * D_ * 2));
    _Float16* ACT16  = (_Float16*)(ws + take((size_t)M_ * FOURD * 2));   // silu(qkvu)
    float*    biasB  = (float*)(ws + take((size_t)B_ * N_ * N_ * 4));
    float*    AV     = (float*)(ws + take((size_t)M_ * D_ * 4));
    _Float16* Y16    = (_Float16*)(ws + take((size_t)M_ * D_ * 2));
    _Float16* VG16   = (_Float16*)(ws + take((size_t)M_ * FOURD * 2));
    _Float16* P16    = (_Float16*)(ws + take((size_t)M_ * TWOD * 2));
    if (o > ws_size) return;

    auto cvt = [&](const float* src, _Float16* dst, size_t n) {
        convert_f16_kernel<<<dim3((unsigned)((n + 255) / 256)), dim3(256), 0, stream>>>(src, dst, (long)n);
    };
    cvt(qkvuW, qkvu16, (size_t)L_ * D_ * FOURD);
    cvt(outW,  out16w, (size_t)L_ * D_ * D_);
    cvt(fc1W,  fc1_16, (size_t)L_ * D_ * FOURD);
    cvt(fc2W,  fc2_16, (size_t)L_ * TWOD * D_);

    const dim3 blk256(256), blk32(32);
    const unsigned gy = (M_ + 127) / 128;

    for (int l = 0; l < L_; ++l) {
        const float* x = (l == 0) ? seqs : out;

        {   // relative attention bias for this layer
            long tot = (long)B_ * N_ * N_;
            bias_kernel<<<dim3((unsigned)((tot + 255) / 256)), blk256, 0, stream>>>(
                ats, tsW + (size_t)l * (NBUCK + 1), posW + (size_t)l * (2 * N_ - 1), biasB);
        }

        // layernorm(x) -> Xn16
        norm_f16_kernel<<<dim3(M_), blk256, 0, stream>>>(x, Xn16, 0);

        // qkvu GEMM + fused silu -> ACT16
        gemm_f16_kernel<<<dim3(FOURD / 128, gy), blk256, 0, stream>>>(
            Xn16, qkvu16 + (size_t)l * D_ * FOURD, ACT16, nullptr, nullptr, nullptr,
            M_, D_, FOURD, 0);

        // fused attention -> AV
        attention_kernel<<<dim3((N_ + 15) / 16, H_, B_), blk32, 0, stream>>>(ACT16, biasB, AV);

        // u * layernorm(av) -> Y16
        uln_kernel<<<dim3(M_), blk256, 0, stream>>>(AV, ACT16, Y16);

        // out projection + out_b + residual(x) -> h (d_out)
        gemm_f16_kernel<<<dim3(D_ / 128, gy), blk256, 0, stream>>>(
            Y16, out16w + (size_t)l * D_ * D_, nullptr, out,
            outBv + (size_t)l * D_, x, M_, D_, D_, 2);

        // rmsnorm(h) -> Xn16
        norm_f16_kernel<<<dim3(M_), blk256, 0, stream>>>(out, Xn16, 1);

        // fc1 GEMM + bias -> VG16
        gemm_f16_kernel<<<dim3(FOURD / 128, gy), blk256, 0, stream>>>(
            Xn16, fc1_16 + (size_t)l * D_ * FOURD, VG16, nullptr,
            fc1Bv + (size_t)l * FOURD, nullptr, M_, D_, FOURD, 1);

        // silu(g) * vv -> P16
        {
            long tot = (long)M_ * TWOD;
            swiglu_kernel<<<dim3((unsigned)((tot + 255) / 256)), blk256, 0, stream>>>(VG16, P16);
        }

        // fc2 GEMM + bias + residual(h) -> h (d_out)
        gemm_f16_kernel<<<dim3(D_ / 128, gy), blk256, 0, stream>>>(
            P16, fc2_16 + (size_t)l * TWOD * D_, nullptr, out,
            fc2Bv + (size_t)l * D_, out, M_, TWOD, D_, 2);
    }
}
